// ngnn_mnist_11192684773705
// MI455X (gfx1250) — matrix operand order repaired and checked by reading, not compiled, not run on hardware
//
#include <hip/hip_runtime.h>
#include <stdint.h>

// ---------------------------------------------------------------------------
// Problem: y = (1 - x @ ternarize(W1)^T / 784) @ W2^T + b2
//   x  [131072 x 784] f32,  W1 [128 x 784] f32, W2 [10 x 128] f32, b2 [10]
// Rewritten: y[b,o] = c[o] - (sum_h P[b,h]*W2[o,h]) / 784
//   where P = x @ Wq^T (f16 WMMA, f32 accum), c[o] = b2[o] + sum_h W2[o,h]
// HBM-bound on x (411 MB); Wq(f16) = 200 KB lives in L2, broadcast via
// async global->LDS double buffering.
// ---------------------------------------------------------------------------

typedef _Float16 v16h __attribute__((ext_vector_type(16)));
typedef float    v8f  __attribute__((ext_vector_type(8)));
typedef float    v4f  __attribute__((ext_vector_type(4)));

#define B_ROWS   131072
#define IN_K     784
#define KSTEPS   25          // 25 * 32 = 800 (K padded, last 16 zero)
#define H_DIM    128
#define OUT_N    10

__device__ __forceinline__ v4f ntload4(const float* p) {
  return __builtin_nontemporal_load((const v4f*)p);
}

// Pack 16 f32 -> v16h in A-fragment per-lane order (i = 0..15)
__device__ __forceinline__ v16h pack16(v4f a, v4f b, v4f c, v4f d) {
  v16h r;
  r[0]=(_Float16)a.x;  r[1]=(_Float16)a.y;  r[2]=(_Float16)a.z;  r[3]=(_Float16)a.w;
  r[4]=(_Float16)b.x;  r[5]=(_Float16)b.y;  r[6]=(_Float16)b.z;  r[7]=(_Float16)b.w;
  r[8]=(_Float16)c.x;  r[9]=(_Float16)c.y;  r[10]=(_Float16)c.z; r[11]=(_Float16)c.w;
  r[12]=(_Float16)d.x; r[13]=(_Float16)d.y; r[14]=(_Float16)d.z; r[15]=(_Float16)d.w;
  return r;
}

__device__ __forceinline__ v8f wmma_f16(v16h a, v16h b, v8f c) {
  return __builtin_amdgcn_wmma_f32_16x16x32_f16(false, a, false, b, (short)0, c,
                                                false, false);
}

// ---------------------------------------------------------------------------
// Prep 1: ternarize W1 into f16 B-fragments, layout:
//   wqfrag[((kk*8 + j)*32 + lane)*16 + i]  (kk = K/32 step, j = H/16 tile)
//   B[k][n] = Wq[n][k]; lane half h holds K=kbase+8h+i for i<8 and
//   K=kbase+16+8h+(i-8) for i>=8 (col n=16j+lane%16). K >= 784 -> 0.
// ---------------------------------------------------------------------------
__global__ void prep_wq(const float* __restrict__ W1, _Float16* __restrict__ wqfrag) {
  int idx = blockIdx.x * 256 + threadIdx.x;
  if (idx >= KSTEPS * 8 * 512) return;
  int i    = idx & 15;
  int lane = (idx >> 4) & 31;
  int frag = idx >> 9;
  int kk   = frag >> 3;
  int j    = frag & 7;
  int n    = 16 * j + (lane & 15);
  int kidx = (i < 8) ? (8 * (lane >> 4) + i) : (16 + 8 * (lane >> 4) + (i - 8));
  int k    = 32 * kk + kidx;
  float v = 0.0f;
  if (k < IN_K) {
    float w = W1[n * IN_K + k];
    v = (w > 0.5f) ? 1.0f : ((w < -0.5f) ? -1.0f : 0.0f);
  }
  wqfrag[idx] = (_Float16)v;
}

// Prep 2: W2 -> f16 B-fragments (4 K-steps over H=128, one N-tile, o>=10 -> 0)
__global__ void prep_w2(const float* __restrict__ W2, _Float16* __restrict__ w2frag) {
  int idx = blockIdx.x * 256 + threadIdx.x;
  if (idx >= 4 * 512) return;
  int i    = idx & 15;
  int lane = (idx >> 4) & 31;
  int kk   = idx >> 9;
  int o    = lane & 15;
  int kidx = (i < 8) ? (8 * (lane >> 4) + i) : (16 + 8 * (lane >> 4) + (i - 8));
  int h    = 32 * kk + kidx;
  float v  = (o < OUT_N) ? W2[o * H_DIM + h] : 0.0f;
  w2frag[idx] = (_Float16)v;
}

// Prep 3: c[o] = b2[o] + sum_h W2[o][h]  (o padded to 16 with zeros)
__global__ void prep_c(const float* __restrict__ W2, const float* __restrict__ b2,
                       float* __restrict__ cvec) {
  int o = threadIdx.x;
  if (o >= 16) return;
  float c = 0.0f;
  if (o < OUT_N) {
    c = b2[o];
    for (int h = 0; h < H_DIM; ++h) c += W2[o * H_DIM + h];
  }
  cvec[o] = c;
}

// ---------------------------------------------------------------------------
// Main fused kernel: 256 threads (8 waves), 256 batch rows per block.
// Each wave: 32 rows x full H=128 strip (16 f32 WMMA accumulators),
// 25 K-steps with async-LDS double-buffered Wq fragments, then fc2 via
// LDS transpose + 4 WMMAs.
// ---------------------------------------------------------------------------
__global__ __launch_bounds__(256, 1) void fused_fwd(
    const float* __restrict__ x,
    const _Float16* __restrict__ wqfrag,
    const _Float16* __restrict__ w2frag,
    const float* __restrict__ cvec,
    float* __restrict__ out)
{
  __shared__ __align__(32) _Float16 sB[2][4096];      // 2 x 8 KB staged Wq frags
  __shared__ __align__(16) float    sP[8][16 * 132];  // per-wave padded P tile

  const int tid  = threadIdx.x;
  const int wave = tid >> 5;
  const int lane = tid & 31;
  const int l15  = lane & 15;
  const int hi   = lane >> 4;
  const size_t rowBase = (size_t)blockIdx.x * 256 + (size_t)wave * 32;

  const float* xrow0 = x + (rowBase + (size_t)l15) * IN_K;  // tile0 row
  const float* xrow1 = xrow0 + (size_t)16 * IN_K;           // tile1 row

  v8f acc0[8], acc1[8];
  #pragma unroll
  for (int j = 0; j < 8; ++j) {
    acc0[j] = (v8f){0,0,0,0,0,0,0,0};
    acc1[j] = (v8f){0,0,0,0,0,0,0,0};
  }

  const unsigned long long wq64 = (unsigned long long)(uintptr_t)wqfrag;
  const unsigned laneoff = (unsigned)(wave * 1024 + lane * 16);  // bytes

  // async copy of K-step s fragment block (8 KB) into LDS buffer `buf`;
  // each wave moves its 1 KB slice as 2 x b128 per lane.
  auto prefetch = [&](int s, int buf) {
    unsigned ldsbase = (unsigned)(uintptr_t)&sB[buf][0];
    unsigned l0 = ldsbase + laneoff;
    unsigned g0 = (unsigned)(s * 8192) + laneoff;
    asm volatile(
        "global_load_async_to_lds_b128 %0, %1, %4\n\t"
        "global_load_async_to_lds_b128 %2, %3, %4"
        :: "v"(l0), "v"(g0), "v"(l0 + 512u), "v"(g0 + 512u), "s"(wq64)
        : "memory");
  };

  prefetch(0, 0);

  #pragma unroll 1
  for (int s = 0; s < KSTEPS; ++s) {
    const int buf = s & 1;
    asm volatile("s_wait_asynccnt 0" ::: "memory");
    __syncthreads();                       // buf ready; other buf free
    if (s + 1 < KSTEPS) prefetch(s + 1, buf ^ 1);

    // A fragments: lane holds row M=l15; lanes>=16 take K offset +8.
    const int kb = 32 * s + hi * 8;
    v4f f0 = ntload4(xrow0 + kb);
    v4f f1 = ntload4(xrow0 + kb + 4);
    v4f g0 = ntload4(xrow1 + kb);
    v4f g1 = ntload4(xrow1 + kb + 4);
    v4f f2 = (v4f){0,0,0,0}, f3 = f2, g2 = f2, g3 = f2;
    if (s < KSTEPS - 1) {                  // K tail 784..799 must be zero
      f2 = ntload4(xrow0 + kb + 16); f3 = ntload4(xrow0 + kb + 20);
      g2 = ntload4(xrow1 + kb + 16); g3 = ntload4(xrow1 + kb + 20);
    }
    v16h a0 = pack16(f0, f1, f2, f3);
    v16h a1 = pack16(g0, g1, g2, g3);

    const _Float16* bb = &sB[buf][lane * 16];
    #pragma unroll
    for (int j = 0; j < 8; ++j) {
      v16h b = *(const v16h*)(bb + j * 512);
      acc0[j] = wmma_f16(a0, b, acc0[j]);
      acc1[j] = wmma_f16(a1, b, acc1[j]);
    }
  }

  // ---- epilogue: fc2 = P[16x128] @ W2^T via LDS transpose + WMMA ----
  float* P = &sP[wave][0];
  const float cv = cvec[l15];

  #pragma unroll 1
  for (int t = 0; t < 2; ++t) {
    const v8f* acc = t ? acc1 : acc0;
    // D layout: lane holds (M = r + 8*hi, N = 16j + l15); row stride 132 floats
    #pragma unroll
    for (int j = 0; j < 8; ++j) {
      #pragma unroll
      for (int r = 0; r < 8; ++r)
        P[(r + 8 * hi) * 132 + 16 * j + l15] = acc[j][r];
    }
    __syncthreads();

    v8f acc2 = (v8f){0,0,0,0,0,0,0,0};
    #pragma unroll
    for (int kk = 0; kk < 4; ++kk) {
      const float* Pr = P + l15 * 132 + 32 * kk + hi * 8;
      v4f p0 = *(const v4f*)(Pr);
      v4f p1 = *(const v4f*)(Pr + 4);
      v4f p2 = *(const v4f*)(Pr + 16);
      v4f p3 = *(const v4f*)(Pr + 20);
      v16h a2 = pack16(p0, p1, p2, p3);
      v16h b2 = *(const v16h*)(w2frag + kk * 512 + lane * 16);
      acc2 = wmma_f16(a2, b2, acc2);
    }
    __syncthreads();

    if (l15 < OUT_N) {
      const size_t row0 = rowBase + (size_t)(16 * t + 8 * hi);
      #pragma unroll
      for (int r = 0; r < 8; ++r) {
        float y = cv - acc2[r] * (1.0f / 784.0f);
        __builtin_nontemporal_store(y, out + (row0 + (size_t)r) * OUT_N + l15);
      }
    }
  }
}

// ---------------------------------------------------------------------------
extern "C" void kernel_launch(void* const* d_in, const int* in_sizes, int n_in,
                              void* d_out, int out_size, void* d_ws, size_t ws_size,
                              hipStream_t stream) {
  const float* x  = (const float*)d_in[0];  // [131072*784]
  const float* W1 = (const float*)d_in[1];  // [128*784]
  const float* W2 = (const float*)d_in[2];  // [10*128]
  const float* b2 = (const float*)d_in[3];  // [10]
  float* out = (float*)d_out;               // [131072*10]

  char* ws = (char*)d_ws;
  _Float16* wqfrag = (_Float16*)(ws);            // 25*8*512 f16 = 204800 B
  _Float16* w2frag = (_Float16*)(ws + 204800);   // 4*512 f16   =   4096 B
  float*    cvec   = (float*)(ws + 208896);      // 16 f32      =     64 B

  prep_wq<<<(KSTEPS * 8 * 512 + 255) / 256, 256, 0, stream>>>(W1, wqfrag);
  prep_w2<<<(4 * 512 + 255) / 256, 256, 0, stream>>>(W2, w2frag);
  prep_c<<<1, 16, 0, stream>>>(W2, b2, cvec);
  fused_fwd<<<B_ROWS / 256, 256, 0, stream>>>(x, wqfrag, w2frag, cvec, out);
}
